// CrossAttentionStudent_74320114090079
// MI455X (gfx1250) — matrix operand order repaired and checked by reading, not compiled, not run on hardware
//
#include <hip/hip_runtime.h>
#include <hip/hip_bf16.h>

typedef __bf16 bf16_t;
typedef __attribute__((ext_vector_type(16))) __bf16 v16bf;
typedef __attribute__((ext_vector_type(8)))  __bf16 v8bf;
typedef __attribute__((ext_vector_type(4)))  __bf16 v4bf;
typedef __attribute__((ext_vector_type(8)))  float  v8f;

#define M_TILE   64
#define THREADS  256   // 8 waves: 4 (M) x 2 (N) partition -> 2 waves/SIMD32
#define NWAVES_M 4
#define NWAVES_N 2

// ============================================================
// Prepass: convert f32 weights -> bf16 in d_ws, zero-padded.
// ============================================================
__global__ void convert_pad(const float* __restrict__ src, bf16_t* __restrict__ dst,
                            int rows, int cols, int cpad, long total) {
  for (long i = (long)blockIdx.x * blockDim.x + threadIdx.x; i < total;
       i += (long)gridDim.x * blockDim.x) {
    int r = (int)(i / cpad), c = (int)(i % cpad);
    float v = (r < rows && c < cols) ? src[(long)r * cols + c] : 0.f;
    dst[i] = (bf16_t)v;
  }
}

// ============================================================
// Fragment loads (contiguous k order: k = 16*(lane/16) + s, wave32)
// ============================================================
__device__ __forceinline__ v16bf lds_load_a(const bf16_t* Arow, int kb) {
  v8bf lo = *reinterpret_cast<const v8bf*>(Arow + kb);
  v8bf hi = *reinterpret_cast<const v8bf*>(Arow + kb + 8);
  v16bf a;
#pragma unroll
  for (int j = 0; j < 8; ++j) { a[j] = lo[j]; a[8 + j] = hi[j]; }
  return a;
}
__device__ __forceinline__ v16bf glob_load_b(const bf16_t* Bp, int kb) {
  return *reinterpret_cast<const v16bf*>(Bp + kb);
}
__device__ __forceinline__ v8f wmma_bf16(v16bf a, v16bf b, v8f c) {
  return __builtin_amdgcn_wmma_f32_16x16x32_bf16(false, a, false, b,
                                                 (short)0, c, false, false);
}

// ============================================================
// GEMM: C = act(A @ W^T + bias); A/C in LDS bf16, W padded bf16 (L2).
// Wave partition: wid&3 over M tiles, wid>>2 over groups of 4 N tiles.
// One A fragment feeds 4 WMMAs.
// ============================================================
__device__ void gemm_lds(const bf16_t* A, int lda, int Mrows,
                         const bf16_t* W, int ldw, int Kpad,
                         int outPad, const float* bias,
                         bf16_t* C, int ldc, bool elu) {
  int wid  = threadIdx.x >> 5;
  int mtw  = wid & (NWAVES_M - 1);
  int ntw  = wid >> 2;
  int lane = threadIdx.x & 31;
  int l15  = lane & 15;
  int hi   = lane >> 4;
  int mtiles = Mrows >> 4, ntiles = outPad >> 4, ktiles = Kpad >> 5;
  for (int mt = mtw; mt < mtiles; mt += NWAVES_M) {
    const bf16_t* Arow = A + (mt * 16 + l15) * lda + hi * 16;
    for (int nt0 = ntw * 4; nt0 < ntiles; nt0 += 4 * NWAVES_N) {
      const bf16_t* Bp = W + (nt0 * 16 + l15) * ldw + hi * 16;
      v8f a0 = {}, a1 = {}, a2 = {}, a3 = {};
#pragma unroll 2
      for (int kt = 0; kt < ktiles; ++kt) {
        int kb = kt * 32;
        v16bf a  = lds_load_a(Arow, kb);
        v16bf b0 = glob_load_b(Bp, kb);
        v16bf b1 = glob_load_b(Bp + 16 * ldw, kb);
        v16bf b2 = glob_load_b(Bp + 32 * ldw, kb);
        v16bf b3 = glob_load_b(Bp + 48 * ldw, kb);
        a0 = wmma_bf16(a, b0, a0);
        a1 = wmma_bf16(a, b1, a1);
        a2 = wmma_bf16(a, b2, a2);
        a3 = wmma_bf16(a, b3, a3);
      }
#pragma unroll
      for (int j = 0; j < 4; ++j) {
        v8f acc = (j == 0) ? a0 : (j == 1) ? a1 : (j == 2) ? a2 : a3;
        int n = (nt0 + j) * 16 + l15;
        float bv = bias[n];
#pragma unroll
        for (int r = 0; r < 8; ++r) {
          float x = acc[r] + bv;
          if (elu) x = (x > 0.f) ? x : (__expf(x) - 1.f);
          C[(mt * 16 + r + hi * 8) * ldc + n] = (bf16_t)x;
        }
      }
    }
  }
}

// Final head: out[row,0:12] = A[64x64] @ Wa1p^T + ba1 (Wa1p padded 16x64).
__device__ void gemm_out(const bf16_t* A, int lda, int Mrows,
                         const bf16_t* W, const float* bias,
                         float* out, int batch, int row0) {
  int wid  = threadIdx.x >> 5;
  int lane = threadIdx.x & 31;
  int l15  = lane & 15;
  int hi   = lane >> 4;
  int mtiles = Mrows >> 4;
  for (int mt = wid; mt < mtiles; mt += THREADS / 32) {
    const bf16_t* Arow = A + (mt * 16 + l15) * lda + hi * 16;
    const bf16_t* Bp   = W + l15 * 64 + hi * 16;
    v8f acc = {};
#pragma unroll
    for (int kt = 0; kt < 2; ++kt) {
      v16bf a = lds_load_a(Arow, kt * 32);
      v16bf b = glob_load_b(Bp, kt * 32);
      acc = wmma_bf16(a, b, acc);
    }
    if (l15 < 12) {
      float bv = bias[l15];
#pragma unroll
      for (int r = 0; r < 8; ++r) {
        int row = row0 + mt * 16 + r + hi * 8;
        if (row < batch) out[(long)row * 12 + l15] = acc[r] + bv;
      }
    }
  }
}

// ============================================================
// Fused forward kernel
// ============================================================
__global__ __launch_bounds__(THREADS)
void cross_attn_student_fused(
    const float* __restrict__ obs,
    const bf16_t* __restrict__ eW0p, const bf16_t* __restrict__ eW1p,
    const bf16_t* __restrict__ eW2p, const bf16_t* __restrict__ We0p,
    const bf16_t* __restrict__ We1p, const bf16_t* __restrict__ Wqp,
    const bf16_t* __restrict__ Wkp,  const bf16_t* __restrict__ Wvp,
    const bf16_t* __restrict__ Wiqp, const bf16_t* __restrict__ Wikp,
    const bf16_t* __restrict__ Wivp, const bf16_t* __restrict__ owp,
    const bf16_t* __restrict__ Wa0p, const bf16_t* __restrict__ Wa1p,
    const float* __restrict__ eb0, const float* __restrict__ eb1,
    const float* __restrict__ eb2, const float* __restrict__ be0,
    const float* __restrict__ be1, const float* __restrict__ bq,
    const float* __restrict__ bk,  const float* __restrict__ bv,
    const float* __restrict__ in_b, const float* __restrict__ ob,
    const float* __restrict__ ba0, const float* __restrict__ ba1,
    float* __restrict__ out, int batch) {
  // 224 KB LDS: activations never leave the WGP.
  __shared__ bf16_t s_obs[M_TILE * 256];      // obs bf16 (K-pad 256); later qh
  __shared__ bf16_t s_x  [M_TILE * 512];      // ping (also f32 staging for async obs copy)
  __shared__ bf16_t s_y  [M_TILE * 512];      // pong; later vh (192x128)
  __shared__ bf16_t s_q  [M_TILE * 128];      // q; later ctx
  __shared__ bf16_t s_lat[3 * M_TILE * 128];  // latents (192x128); later kh

  int row0 = blockIdx.x * M_TILE;
  int tid  = threadIdx.x;

  // ---- Stage obs via CDNA5 async global->LDS (b32: rows only 4B aligned),
  //      then vectorized LDS f32 -> LDS bf16 convert with zero K-padding.
  {
    float* stage = reinterpret_cast<float*>(s_x);            // 64 x 236 f32
    unsigned long long gbase = (unsigned long long)(obs + (long)row0 * 235);
    for (int idx = tid; idx < M_TILE * 256; idx += THREADS) {
      int r = idx >> 8, c = idx & 255;
      if (c < 235 && row0 + r < batch) {
        unsigned lds = (unsigned)(uintptr_t)(stage + r * 236 + c);
        unsigned gof = (unsigned)((r * 235 + c) * 4);
        asm volatile("global_load_async_to_lds_b32 %0, %1, %2"
                     :: "v"(lds), "v"(gof), "s"(gbase) : "memory");
      }
    }
    asm volatile("s_wait_asynccnt 0x0" ::: "memory");
    __syncthreads();
    for (int idx = tid; idx < M_TILE * 64; idx += THREADS) { // 4-dword chunks
      int r = idx >> 6, c4 = idx & 63, cb = c4 * 4;
      float4 v = make_float4(0.f, 0.f, 0.f, 0.f);
      if (c4 < 59) v = *reinterpret_cast<const float4*>(stage + r * 236 + cb);
      v4bf o;
      o[0] = (bf16_t)((cb + 0 < 235) ? v.x : 0.f);
      o[1] = (bf16_t)((cb + 1 < 235) ? v.y : 0.f);
      o[2] = (bf16_t)((cb + 2 < 235) ? v.z : 0.f);
      o[3] = (bf16_t)((cb + 3 < 235) ? v.w : 0.f);
      *reinterpret_cast<v4bf*>(&s_obs[r * 256 + cb]) = o;
    }
    __syncthreads();
  }

  // ---- Query path: obs -> 256 -> 128 -> q
  gemm_lds(s_obs, 256, M_TILE, We0p, 256, 256, 256, be0, s_x, 256, true);
  __syncthreads();
  gemm_lds(s_x, 256, M_TILE, We1p, 256, 256, 128, be1, s_y, 128, true);
  __syncthreads();
  gemm_lds(s_y, 128, M_TILE, Wqp, 128, 128, 128, bq, s_q, 128, false);
  __syncthreads();

  // ---- Expert encoders: obs -> 512 -> 256 -> latent(128), 3 experts
  for (int e = 0; e < 3; ++e) {
    gemm_lds(s_obs, 256, M_TILE, eW0p + (long)e * 512 * 256, 256, 256,
             512, eb0 + e * 512, s_x, 512, true);
    __syncthreads();
    gemm_lds(s_x, 512, M_TILE, eW1p + (long)e * 256 * 512, 512, 512,
             256, eb1 + e * 256, s_y, 256, true);
    __syncthreads();
    gemm_lds(s_y, 256, M_TILE, eW2p + (long)e * 128 * 256, 256, 256,
             128, eb2 + e * 128, s_lat + e * M_TILE * 128, 128, true);
    __syncthreads();
  }

  // ---- k/v + head projections, batched over 3 experts (192 rows)
  gemm_lds(s_lat, 128, 3 * M_TILE, Wvp, 128, 128, 128, bv, s_x, 128, false);
  __syncthreads();
  gemm_lds(s_x, 128, 3 * M_TILE, Wivp, 128, 128, 128, in_b + 256, s_y, 128, false); // vh
  __syncthreads();
  gemm_lds(s_lat, 128, 3 * M_TILE, Wkp, 128, 128, 128, bk, s_x, 128, false);
  __syncthreads();
  gemm_lds(s_x, 128, 3 * M_TILE, Wikp, 128, 128, 128, in_b + 128, s_lat, 128, false); // kh
  __syncthreads();
  gemm_lds(s_q, 128, M_TILE, Wiqp, 128, 128, 128, in_b, s_obs, 128, false);          // qh
  __syncthreads();

  // ---- 4-head, 3-expert softmax attention: thread = (row, head), ctx -> s_q
  {
    int r = tid & 63;
    int h = tid >> 6;            // 0..3
    const float scale = 0.17677669529663687f; // 1/sqrt(32)
    float s0 = 0.f, s1 = 0.f, s2 = 0.f;
#pragma unroll 8
    for (int d = 0; d < 32; ++d) {
      float qv = (float)s_obs[r * 128 + h * 32 + d];
      s0 += qv * (float)s_lat[(0 * M_TILE + r) * 128 + h * 32 + d];
      s1 += qv * (float)s_lat[(1 * M_TILE + r) * 128 + h * 32 + d];
      s2 += qv * (float)s_lat[(2 * M_TILE + r) * 128 + h * 32 + d];
    }
    s0 *= scale; s1 *= scale; s2 *= scale;
    float m  = fmaxf(s0, fmaxf(s1, s2));
    float e0 = __expf(s0 - m), e1 = __expf(s1 - m), e2 = __expf(s2 - m);
    float inv = 1.f / (e0 + e1 + e2);
    float p0 = e0 * inv, p1 = e1 * inv, p2 = e2 * inv;
#pragma unroll 8
    for (int d = 0; d < 32; ++d) {
      float c = p0 * (float)s_y[(0 * M_TILE + r) * 128 + h * 32 + d]
              + p1 * (float)s_y[(1 * M_TILE + r) * 128 + h * 32 + d]
              + p2 * (float)s_y[(2 * M_TILE + r) * 128 + h * 32 + d];
      s_q[r * 128 + h * 32 + d] = (bf16_t)c;
    }
  }
  __syncthreads();

  // ---- Output head: ctx -> attended -> 64 (ELU) -> 12
  gemm_lds(s_q, 128, M_TILE, owp, 128, 128, 128, ob, s_x, 128, false);
  __syncthreads();
  gemm_lds(s_x, 128, M_TILE, Wa0p, 128, 128, 64, ba0, s_y, 64, true);
  __syncthreads();
  gemm_out(s_y, 64, M_TILE, Wa1p, ba1, out, batch, row0);
}

// ============================================================
extern "C" void kernel_launch(void* const* d_in, const int* in_sizes, int n_in,
                              void* d_out, int out_size, void* d_ws, size_t ws_size,
                              hipStream_t stream) {
  (void)n_in; (void)ws_size; (void)out_size;
  const float* obs  = (const float*)d_in[0];
  const float* eW0  = (const float*)d_in[1];
  const float* eb0  = (const float*)d_in[2];
  const float* eW1  = (const float*)d_in[3];
  const float* eb1  = (const float*)d_in[4];
  const float* eW2  = (const float*)d_in[5];
  const float* eb2  = (const float*)d_in[6];
  const float* We0  = (const float*)d_in[7];
  const float* be0  = (const float*)d_in[8];
  const float* We1  = (const float*)d_in[9];
  const float* be1  = (const float*)d_in[10];
  const float* Wq   = (const float*)d_in[11];
  const float* bq   = (const float*)d_in[12];
  const float* Wk   = (const float*)d_in[13];
  const float* bk   = (const float*)d_in[14];
  const float* Wv   = (const float*)d_in[15];
  const float* bv   = (const float*)d_in[16];
  const float* in_w = (const float*)d_in[17];
  const float* in_b = (const float*)d_in[18];
  const float* ow   = (const float*)d_in[19];
  const float* ob   = (const float*)d_in[20];
  const float* Wa0  = (const float*)d_in[21];
  const float* ba0  = (const float*)d_in[22];
  const float* Wa1  = (const float*)d_in[23];
  const float* ba1  = (const float*)d_in[24];

  // bf16 padded weight arena in d_ws (~2.2 MB)
  bf16_t* ws = (bf16_t*)d_ws;
  size_t off = 0;
  auto take = [&](size_t n) { bf16_t* p = ws + off; off += n; return p; };
  bf16_t* eW0p = take((size_t)3 * 512 * 256);
  bf16_t* eW1p = take((size_t)3 * 256 * 512);
  bf16_t* eW2p = take((size_t)3 * 128 * 256);
  bf16_t* We0p = take((size_t)256 * 256);
  bf16_t* We1p = take((size_t)128 * 256);
  bf16_t* Wqp  = take((size_t)128 * 128);
  bf16_t* Wkp  = take((size_t)128 * 128);
  bf16_t* Wvp  = take((size_t)128 * 128);
  bf16_t* Wiqp = take((size_t)128 * 128);
  bf16_t* Wikp = take((size_t)128 * 128);
  bf16_t* Wivp = take((size_t)128 * 128);
  bf16_t* owp  = take((size_t)128 * 128);
  bf16_t* Wa0p = take((size_t)64 * 128);
  bf16_t* Wa1p = take((size_t)16 * 64);

  auto conv = [&](const float* s, bf16_t* d, int rows, int cols, int cpad, int rpad) {
    long total = (long)rpad * cpad;
    int grid = (int)((total + 255) / 256);
    convert_pad<<<grid, 256, 0, stream>>>(s, d, rows, cols, cpad, total);
  };
  conv(eW0, eW0p, 3 * 512, 235, 256, 3 * 512);
  conv(eW1, eW1p, 3 * 256, 512, 512, 3 * 256);
  conv(eW2, eW2p, 3 * 128, 256, 256, 3 * 128);
  conv(We0, We0p, 256, 235, 256, 256);
  conv(We1, We1p, 128, 256, 256, 128);
  conv(Wq, Wqp, 128, 128, 128, 128);
  conv(Wk, Wkp, 128, 128, 128, 128);
  conv(Wv, Wvp, 128, 128, 128, 128);
  conv(in_w,                 Wiqp, 128, 128, 128, 128);
  conv(in_w + 128 * 128,     Wikp, 128, 128, 128, 128);
  conv(in_w + 2 * 128 * 128, Wivp, 128, 128, 128, 128);
  conv(ow, owp, 128, 128, 128, 128);
  conv(Wa0, Wa0p, 64, 128, 128, 64);
  conv(Wa1, Wa1p, 12, 64, 64, 16);

  int batch = in_sizes[0] / 235;
  int grid  = (batch + M_TILE - 1) / M_TILE;
  cross_attn_student_fused<<<grid, THREADS, 0, stream>>>(
      obs, eW0p, eW1p, eW2p, We0p, We1p, Wqp, Wkp, Wvp, Wiqp, Wikp, Wivp,
      owp, Wa0p, Wa1p,
      eb0, eb1, eb2, be0, be1, bq, bk, bv, in_b, ob, ba0, ba1,
      (float*)d_out, batch);
}
